// DualLossDynamicMHA_52527450030304
// MI455X (gfx1250) — matrix operand order repaired and checked by reading, not compiled, not run on hardware
//
#include <hip/hip_runtime.h>

// ---------------------------------------------------------------------------
// Types matching gfx1250 WMMA builtins
// ---------------------------------------------------------------------------
typedef __bf16 bf16_t;
typedef __bf16 v16bf __attribute__((ext_vector_type(16)));
typedef __bf16 v8bf  __attribute__((ext_vector_type(8)));
typedef __bf16 v4bf  __attribute__((ext_vector_type(4)));
typedef float  v8f   __attribute__((ext_vector_type(8)));

#define D_MODEL   512
#define NUM_HEADS 8
#define MIN_DIM   9
#define B_SZ      4
#define T_SZ      2048
#define DPMAX     512   // max padded head dim (row stride in LDS)

// ---------------------------------------------------------------------------
// Fragment loaders (CDNA5 ISA §7.12.2 layouts, wave32)
// A-matrix 16x32 bf16: lane<16 holds K{0..7,16..23}; lane>=16 holds K{8..15,24..31}
// B-matrix 32x16 bf16: lane holds col N=lane%16; lane<16 holds K{0..7,16..23}; lane>=16 holds K{8..15,24..31}
// ---------------------------------------------------------------------------
__device__ __forceinline__ v16bf load_a_frag(const bf16_t* base, int stride,
                                             int row, int kbase, int lane) {
    const int half = (lane >> 4) * 8;
    union { v16bf v; v8bf h[2]; } u;
    u.h[0] = *(const v8bf*)(base + (size_t)row * stride + kbase + half);
    u.h[1] = *(const v8bf*)(base + (size_t)row * stride + kbase + 16 + half);
    return u.v;
}

__device__ __forceinline__ v16bf load_b_frag(const bf16_t* base, int stride,
                                             int col, int kbase, int lane) {
    const int half = (lane >> 4) * 8;
    union { v16bf v; v8bf h[2]; } u;
    u.h[0] = *(const v8bf*)(base + (size_t)col * stride + kbase + half);
    u.h[1] = *(const v8bf*)(base + (size_t)col * stride + kbase + 16 + half);
    return u.v;
}

__device__ __forceinline__ void store_bf4(bf16_t* p, float4 v) {
    v4bf t;
    t.x = (bf16_t)v.x; t.y = (bf16_t)v.y; t.z = (bf16_t)v.z; t.w = (bf16_t)v.w;
    *(v4bf*)p = t;
}

// ---------------------------------------------------------------------------
// Kernel 1: integer head-dim partition (replicates numpy floor + residual fix)
// meta[0..7] = dims, meta[8..15] = prefix offsets
// ---------------------------------------------------------------------------
__global__ void head_dims_kernel(const float* __restrict__ logits,
                                 int* __restrict__ meta) {
    if (threadIdx.x != 0 || blockIdx.x != 0) return;
    float l[NUM_HEADS], mx = -1e30f;
    for (int i = 0; i < NUM_HEADS; ++i) { l[i] = logits[i]; mx = fmaxf(mx, l[i]); }
    float e[NUM_HEADS], s = 0.f;
    for (int i = 0; i < NUM_HEADS; ++i) { e[i] = expf(l[i] - mx); s += e[i]; }
    int   di[NUM_HEADS]; float res[NUM_HEADS]; int sum = 0;
    const float span = (float)(D_MODEL - MIN_DIM * NUM_HEADS);
    for (int i = 0; i < NUM_HEADS; ++i) {
        float df = (float)MIN_DIM + (e[i] / s) * span;
        float fl = floorf(df);
        di[i]  = (int)fl;
        res[i] = df - fl;
        sum   += di[i];
    }
    int diff = D_MODEL - sum;
    bool used[NUM_HEADS] = {false};
    if (diff > 0) {
        for (int k = 0; k < diff; ++k) {
            int best = -1; float bv = -1.f;
            for (int i = 0; i < NUM_HEADS; ++i)
                if (!used[i] && res[i] > bv) { bv = res[i]; best = i; }
            used[best] = true; di[best] += 1;
        }
    } else if (diff < 0) {
        for (int k = 0; k < -diff; ++k) {
            int best = -1; float bv = 1e30f;
            for (int i = 0; i < NUM_HEADS; ++i)
                if (!used[i] && res[i] < bv) { bv = res[i]; best = i; }
            used[best] = true; di[best] -= 1;
        }
    }
    int o = 0;
    for (int i = 0; i < NUM_HEADS; ++i) { meta[i] = di[i]; meta[8 + i] = o; o += di[i]; }
}

// ---------------------------------------------------------------------------
// Kernel 2/4: C[M,N] = A[M,K] * W[N,K]^T + bias[N]   (bf16 WMMA, f32 accum)
// block tile 128x128, 8 waves in 4(M)x2(N), wave tile 32x64 (2x4 WMMA frags)
// Staging: float4 (global_load_b128) batched 8-deep, ds_store_b64 bf16 packs.
// ---------------------------------------------------------------------------
__global__ __launch_bounds__(256)
void wmma_gemm_bias(const float* __restrict__ A, const float* __restrict__ W,
                    const float* __restrict__ bias, float* __restrict__ C,
                    int M, int N, int K) {
    __shared__ bf16_t sA[128 * 32];
    __shared__ bf16_t sB[128 * 32];

    const int tid     = threadIdx.x;
    const int lane    = tid & 31;
    const int wid     = tid >> 5;
    const int rowBase = blockIdx.y * 128;
    const int colBase = blockIdx.x * 128;
    const int waveM   = (wid & 3) * 32;
    const int waveN   = (wid >> 2) * 64;

    v8f acc[2][4];
    for (int i = 0; i < 2; ++i)
        for (int j = 0; j < 4; ++j)
            acc[i][j] = v8f{};

    // Each thread stages 4 float4 of A and 4 float4 of B per k-step.
    // Tile = 128 rows x 8 float4; idx = tid + t*256 over 1024 float4 slots.
    for (int k0 = 0; k0 < K; k0 += 32) {
        float4 ra[4], rb[4];
#pragma unroll
        for (int t = 0; t < 4; ++t) {
            int idx = tid + t * 256;
            int r = idx >> 3, c4 = idx & 7;
            ra[t] = *(const float4*)(A + (size_t)(rowBase + r) * K + k0 + c4 * 4);
            rb[t] = *(const float4*)(W + (size_t)(colBase + r) * K + k0 + c4 * 4);
        }
        if (k0 + 32 < K)
            __builtin_prefetch(A + (size_t)(rowBase + (tid >> 1)) * K + k0 + 32, 0, 0);
#pragma unroll
        for (int t = 0; t < 4; ++t) {
            int idx = tid + t * 256;
            int r = idx >> 3, c4 = idx & 7;
            store_bf4(sA + r * 32 + c4 * 4, ra[t]);
            store_bf4(sB + r * 32 + c4 * 4, rb[t]);
        }
        __syncthreads();

        v16bf af[2], bfr[4];
        af[0]  = load_a_frag(sA, 32, waveM + (lane & 15),      0, lane);
        af[1]  = load_a_frag(sA, 32, waveM + 16 + (lane & 15), 0, lane);
#pragma unroll
        for (int j = 0; j < 4; ++j)
            bfr[j] = load_b_frag(sB, 32, waveN + j * 16 + (lane & 15), 0, lane);
#pragma unroll
        for (int i = 0; i < 2; ++i)
#pragma unroll
            for (int j = 0; j < 4; ++j)
                acc[i][j] = __builtin_amdgcn_wmma_f32_16x16x32_bf16(
                    false, af[i], false, bfr[j], (short)0, acc[i][j], false, false);
        __syncthreads();
    }

    // Epilogue: C layout: row = v + 8*(lane/16), col = lane%16
    const int rowAdd = (lane >> 4) * 8;
    const int nIdx   = lane & 15;
    for (int i = 0; i < 2; ++i)
        for (int j = 0; j < 4; ++j) {
            int col = colBase + waveN + j * 16 + nIdx;
            float bv = bias[col];
            for (int v = 0; v < 8; ++v) {
                int row = rowBase + waveM + i * 16 + v + rowAdd;
                C[(size_t)row * N + col] = acc[i][j][v] + bv;
            }
        }
}

// ---------------------------------------------------------------------------
// Kernel 3: flash attention, one wave per (b, h, 16-query-row tile).
// 32 keys per tile; online-softmax rescale folded into WMMA C-frag load.
// ---------------------------------------------------------------------------
extern __shared__ unsigned char smem_raw[];

__global__ __launch_bounds__(32)
void attn_kernel(const float* __restrict__ qkv, const int* __restrict__ meta,
                 float* __restrict__ attn_out) {
    const int lane = threadIdx.x;
    const int bh   = blockIdx.x;
    const int b    = bh >> 3;
    const int h    = bh & 7;
    const int qt   = blockIdx.y;

    const int d    = meta[h];
    const int off  = meta[8 + h];
    const int kd32 = (d + 31) >> 5;   // #32-wide K-steps over head dim
    const int dp32 = kd32 * 32;       // padded head dim
    const int td   = dp32 >> 4;       // #16-wide d tiles for output
    const float scale = rsqrtf((float)d);

    // LDS carve-up
    bf16_t* sQ  = (bf16_t*)smem_raw;          // 16 x DPMAX
    bf16_t* sK  = sQ  + 16 * DPMAX;           // 32 x DPMAX
    bf16_t* sVt = sK  + 32 * DPMAX;           // DPMAX x 32 (transposed V)
    bf16_t* sP  = sVt + DPMAX * 32;           // 16 x 32 probs
    float*  sS  = (float*)(sP + 16 * 32);     // 16 x 32 raw scores
    float*  sO  = sS + 16 * 32;               // 16 x DPMAX output accumulator
    float*  sM  = sO + 16 * DPMAX;            // 16 row maxes
    float*  sL  = sM + 16;                    // 16 row sums
    float*  sAl = sL + 16;                    // 16 per-tile row rescales

    // ---- Stage Q (batched loads, fp32 -> bf16, zero-pad) + init O ----
    for (int r = 0; r < 16; ++r) {
        const float* src = qkv + (size_t)(b * T_SZ + qt * 16 + r) * (3 * D_MODEL) + off;
        for (int c0 = 0; c0 < dp32; c0 += 256) {
            float rq[8];
#pragma unroll
            for (int u = 0; u < 8; ++u) {
                int c = c0 + lane + u * 32;
                rq[u] = (c < d) ? src[c] : 0.f;
            }
#pragma unroll
            for (int u = 0; u < 8; ++u) {
                int c = c0 + lane + u * 32;
                if (c < dp32) {
                    sQ[r * DPMAX + c] = (bf16_t)rq[u];
                    sO[r * DPMAX + c] = 0.f;
                }
            }
        }
    }
    if (lane < 16) { sM[lane] = -1e30f; sL[lane] = 0.f; }
    __syncthreads();

    const int rowAdd = (lane >> 4) * 8;
    const int nIdx   = lane & 15;

    for (int kt = 0; kt < T_SZ / 32; ++kt) {
        // ---- Stage K tile [32 keys x dp32] and V transposed [dp32 x 32 keys] ----
        for (int r = 0; r < 32; ++r) {
            const float* srcr = qkv + (size_t)(b * T_SZ + kt * 32 + r) * (3 * D_MODEL) + off;
            for (int c0 = 0; c0 < dp32; c0 += 256) {
                float rk[8], rv[8];
#pragma unroll
                for (int u = 0; u < 8; ++u) {
                    int c = c0 + lane + u * 32;
                    bool ok = (c < d);
                    rk[u] = ok ? srcr[D_MODEL + c]     : 0.f;
                    rv[u] = ok ? srcr[2 * D_MODEL + c] : 0.f;
                }
#pragma unroll
                for (int u = 0; u < 8; ++u) {
                    int c = c0 + lane + u * 32;
                    if (c < dp32) {
                        sK[r * DPMAX + c] = (bf16_t)rk[u];
                        sVt[c * 32 + r]   = (bf16_t)rv[u];
                    }
                }
            }
        }
        __syncthreads();

        // ---- S = Q * K^T  (16x32, accumulate over head dim) ----
        v8f s0 = v8f{}, s1 = v8f{};
        for (int ks = 0; ks < kd32; ++ks) {
            v16bf a  = load_a_frag(sQ, DPMAX, nIdx, ks * 32, lane);
            v16bf b0 = load_b_frag(sK, DPMAX, nIdx,      ks * 32, lane);
            v16bf b1 = load_b_frag(sK, DPMAX, 16 + nIdx, ks * 32, lane);
            s0 = __builtin_amdgcn_wmma_f32_16x16x32_bf16(
                     false, a, false, b0, (short)0, s0, false, false);
            s1 = __builtin_amdgcn_wmma_f32_16x16x32_bf16(
                     false, a, false, b1, (short)0, s1, false, false);
        }
#pragma unroll
        for (int v = 0; v < 8; ++v) {
            sS[(v + rowAdd) * 32 + nIdx]      = s0[v] * scale;
            sS[(v + rowAdd) * 32 + 16 + nIdx] = s1[v] * scale;
        }
        __syncthreads();

        // ---- Online softmax: lanes 0..15 each own one query row ----
        if (lane < 16) {
            float mold = sM[lane];
            float mx = mold;
            for (int j = 0; j < 32; ++j) mx = fmaxf(mx, sS[lane * 32 + j]);
            float alpha = expf(mold - mx);
            float rs = 0.f;
            for (int j = 0; j < 32; ++j) {
                float p = expf(sS[lane * 32 + j] - mx);
                rs += p;
                sP[lane * 32 + j] = (bf16_t)p;
            }
            sL[lane]  = sL[lane] * alpha + rs;
            sM[lane]  = mx;
            sAl[lane] = alpha;
        }
        __syncthreads();

        // ---- O = alpha*O + P*V  (rescale folded into C-frag load) ----
        float a8[8];
#pragma unroll
        for (int v = 0; v < 8; ++v) a8[v] = sAl[v + rowAdd];
        v16bf pf = load_a_frag(sP, 32, nIdx, 0, lane);
        for (int j = 0; j < td; ++j) {
            v8f o;
#pragma unroll
            for (int v = 0; v < 8; ++v)
                o[v] = sO[(v + rowAdd) * DPMAX + j * 16 + nIdx] * a8[v];
            v16bf bb = load_b_frag(sVt, 32, j * 16 + nIdx, 0, lane);
            o = __builtin_amdgcn_wmma_f32_16x16x32_bf16(
                    false, pf, false, bb, (short)0, o, false, false);
#pragma unroll
            for (int v = 0; v < 8; ++v)
                sO[(v + rowAdd) * DPMAX + j * 16 + nIdx] = o[v];
        }
        __syncthreads();
    }

    // ---- Normalize and write (only valid d columns) ----
    for (int idx = lane; idx < 16 * d; idx += 32) {
        int r = idx / d, c = idx % d;
        attn_out[(size_t)(b * T_SZ + qt * 16 + r) * D_MODEL + off + c] =
            sO[r * DPMAX + c] / sL[r];
    }
}

// ---------------------------------------------------------------------------
// Launch
// ---------------------------------------------------------------------------
extern "C" void kernel_launch(void* const* d_in, const int* in_sizes, int n_in,
                              void* d_out, int out_size, void* d_ws, size_t ws_size,
                              hipStream_t stream) {
    (void)in_sizes; (void)n_in; (void)out_size; (void)ws_size;
    const float* query  = (const float*)d_in[0];
    const float* logits = (const float*)d_in[1];
    const float* qkv_w  = (const float*)d_in[2];
    const float* qkv_b  = (const float*)d_in[3];
    const float* out_w  = (const float*)d_in[4];
    const float* out_b  = (const float*)d_in[5];
    float* out = (float*)d_out;

    unsigned char* ws = (unsigned char*)d_ws;
    int*   meta = (int*)ws;
    float* qkv  = (float*)(ws + 256);
    float* attn = (float*)(ws + 256 + (size_t)B_SZ * T_SZ * 3 * D_MODEL * sizeof(float));

    const int M = B_SZ * T_SZ;

    head_dims_kernel<<<1, 32, 0, stream>>>(logits, meta);

    dim3 g1((3 * D_MODEL) / 128, M / 128);
    wmma_gemm_bias<<<g1, 256, 0, stream>>>(query, qkv_w, qkv_b, qkv,
                                           M, 3 * D_MODEL, D_MODEL);

    size_t shmem = (size_t)(16 * DPMAX + 32 * DPMAX + DPMAX * 32 + 16 * 32) * sizeof(bf16_t)
                 + (size_t)(16 * 32 + 16 * DPMAX + 48) * sizeof(float);
    dim3 g2(B_SZ * NUM_HEADS, T_SZ / 16);
    attn_kernel<<<g2, 32, shmem, stream>>>(qkv, meta, attn);

    dim3 g3(D_MODEL / 128, M / 128);
    wmma_gemm_bias<<<g3, 256, 0, stream>>>(attn, out_w, out_b, out,
                                           M, D_MODEL, D_MODEL);
}
